// AllAtomE3Encoder_62423054680084
// MI455X (gfx1250) — matrix operand order repaired and checked by reading, not compiled, not run on hardware
//
#include <hip/hip_runtime.h>
#include <hip/hip_bf16.h>

typedef __attribute__((ext_vector_type(16))) _Float16 v16h;
typedef __attribute__((ext_vector_type(8)))  _Float16 v8h;
typedef __attribute__((ext_vector_type(8)))  float    v8f;

// ---------------- problem constants ----------------
constexpr int cA   = 45056;          // atoms
constexpr int cE   = 720896;         // edges (= cA * 16)
constexpr int cN   = 2048;           // residues
constexpr int cH   = 128;            // hidden
constexpr int cLAT = 32;

// ---------------- workspace layout (bytes) ----------------
constexpr size_t AL(size_t x) { return (x + 255) & ~(size_t)255; }
constexpr size_t OFF_H32    = 0;
constexpr size_t OFF_HF16   = AL(OFF_H32    + (size_t)cA * cH * 4);
constexpr size_t OFF_AGG    = AL(OFF_HF16   + (size_t)cA * cH * 2);
constexpr size_t OFF_EATTR  = AL(OFF_AGG    + (size_t)cA * cH * 4);   // E x 32 halves (16 rbf + 16 zero pad)
constexpr size_t OFF_Q32    = AL(OFF_EATTR  + (size_t)cE * 32 * 2);
constexpr size_t OFF_SCORES = AL(OFF_Q32    + (size_t)cN * cH * 4);
constexpr size_t OFF_SMAX   = AL(OFF_SCORES + (size_t)cA * 4);
constexpr size_t OFF_DENOM  = AL(OFF_SMAX   + (size_t)cN * 4);
constexpr size_t OFF_POOLED = AL(OFF_DENOM  + (size_t)cN * 4);
constexpr size_t OFF_WFRAG  = AL(OFF_POOLED + (size_t)cN * cH * 4);
// weight-fragment sub-offsets (in halves)
constexpr size_t WF_WE1 = 0;                       // 3 * 9*8*512
constexpr size_t WF_WE2 = WF_WE1 + 3 * 36864;      // 3 * 4*8*512
constexpr size_t WF_WH1 = WF_WE2 + 3 * 16384;      // 3 * 8*8*512
constexpr size_t WF_WH2 = WF_WH1 + 3 * 32768;      // 3 * 4*8*512
constexpr size_t WF_WK  = WF_WH2 + 3 * 16384;      // 4*8*512
constexpr size_t WF_WV  = WF_WK  + 16384;

// ---------------- helpers ----------------
__device__ inline float silu_f(float x) { return x / (1.0f + __expf(-x)); }

__device__ inline v8f v8f_zero() {
    v8f z = {0.f, 0.f, 0.f, 0.f, 0.f, 0.f, 0.f, 0.f};
    return z;
}

// Build a 16x32 f16 A-fragment for this lane from a row pointer.
// CDNA5 layout: lanes 0-15 hold K {0..7,16..23}; lanes 16-31 hold K {8..15,24..31}.
__device__ inline v16h load_a_frag(const _Float16* __restrict__ p, int hi) {
    v8h c0 = *(const v8h*)(p + hi * 8);
    v8h c1 = *(const v8h*)(p + 16 + hi * 8);
    v16h a;
#pragma unroll
    for (int i = 0; i < 8; ++i) { a[i] = c0[i]; a[8 + i] = c1[i]; }
    return a;
}

__device__ inline void atomicMaxF(float* addr, float val) {
    unsigned int* ua = (unsigned int*)addr;
    unsigned int old = *ua;
    while (__uint_as_float(old) < val) {
        unsigned int assumed = old;
        old = atomicCAS(ua, assumed, __float_as_uint(val));
        if (old == assumed) break;
    }
}

// gfx1250 async global->LDS copy (16B per lane), tracked by ASYNCcnt.
__device__ inline void async_copy_b128(const _Float16* lds_dst, const _Float16* gsrc) {
    unsigned int       lds  = (unsigned int)(unsigned long long)(uintptr_t)lds_dst; // low 32 bits = LDS offset
    unsigned long long addr = (unsigned long long)(uintptr_t)gsrc;
    asm volatile("global_load_async_to_lds_b128 %0, %1, off"
                 :: "v"(lds), "v"(addr)
                 : "memory");
}
__device__ inline void async_wait0() {
    asm volatile("s_wait_asynccnt 0" ::: "memory");
}

// 8 distinct B fragments for one K-step, then 8 back-to-back WMMAs:
// one s_wait_dscnt per K-step instead of one per WMMA.
__device__ inline void wmma_kstep(const _Float16* __restrict__ bbase, const v16h& a, v8f acc[8]) {
    v16h b[8];
#pragma unroll
    for (int nt = 0; nt < 8; ++nt) b[nt] = *(const v16h*)(bbase + nt * 512);
#pragma unroll
    for (int nt = 0; nt < 8; ++nt)
        acc[nt] = __builtin_amdgcn_wmma_f32_16x16x32_f16(
            false, a, false, b[nt], (short)0, acc[nt], false, false);
}

// ---------------- weight repack: (K,128) f32 row-major -> B fragments ----------------
// frag (kt,nt) stored as 512 halves: [lane 0..31][i 0..15];
// element (lane,i): n = nt*16 + (lane&15), k = kt*32 + (lane<16 ? 0 : 8) + (i&7) + (i<8 ? 0 : 16)
__global__ void repack_kernel(const float* __restrict__ W, _Float16* __restrict__ dst,
                              int Kdim, int KT) {
    int tid = blockIdx.x * blockDim.x + threadIdx.x;
    int total = KT * 8 * 512;
    if (tid >= total) return;
    int frag   = tid >> 9;
    int within = tid & 511;
    int lane = within >> 4;
    int i    = within & 15;
    int kt = frag >> 3;
    int nt = frag & 7;
    int n = nt * 16 + (lane & 15);
    int k = kt * 32 + ((lane >> 4) << 3) + (i & 7) + ((i >> 3) << 4);
    float v = (k < Kdim) ? W[(size_t)k * cH + n] : 0.f;
    dst[tid] = (_Float16)v;
}

// ---------------- small prep kernels ----------------
__global__ void fill_kernel(float* __restrict__ p, float v, int n) {
    int i = blockIdx.x * blockDim.x + threadIdx.x;
    if (i < n) p[i] = v;
}

__global__ void rbf_kernel(const float* __restrict__ coords,
                           const int* __restrict__ esrc, const int* __restrict__ edst,
                           _Float16* __restrict__ eattr) {
    int e = blockIdx.x * blockDim.x + threadIdx.x;   // grid exact: E % 256 == 0
    int s = esrc[e], d = edst[e];
    float dx = coords[3 * s + 0] - coords[3 * d + 0];
    float dy = coords[3 * s + 1] - coords[3 * d + 1];
    float dz = coords[3 * s + 2] - coords[3 * d + 2];
    float dist = sqrtf(dx * dx + dy * dy + dz * dz);
    const float gamma = (16.f / 5.f) * (16.f / 5.f);   // (R/CUTOFF)^2
    _Float16* row = eattr + (size_t)e * 32;
#pragma unroll
    for (int i = 0; i < 16; ++i) {
        float c = (5.f / 15.f) * (float)i;             // linspace(0, 5, 16)
        float t = dist - c;
        row[i] = (_Float16)__expf(-gamma * t * t);
    }
#pragma unroll
    for (int i = 16; i < 32; ++i) row[i] = (_Float16)0.f;   // K padding
}

__global__ void hinit_kernel(const int* __restrict__ at, const int* __restrict__ ri,
                             const int* __restrict__ rt,
                             const float* __restrict__ aemb, const float* __restrict__ remb,
                             float* __restrict__ h32, _Float16* __restrict__ hf16) {
    int idx = blockIdx.x * blockDim.x + threadIdx.x;   // A*H threads exact
    int a = idx >> 7, c = idx & 127;
    float v = aemb[(size_t)at[a] * cH + c] + remb[(size_t)rt[ri[a]] * cH + c];
    h32[idx] = v;
    hf16[idx] = (_Float16)v;
}

// ---------------- edge message MLP ----------------
// Weights LDS-resident per block (async-staged); each wave grid-strides over
// 16-edge tiles: stage1 WMMA -> SiLU -> LDS -> stage2 WMMA -> SiLU -> atomic scatter.
constexpr int EDGE_GRID = 512;     // 512 blocks * 8 waves = 4096 waves; 45056 tiles = 11 per wave
__global__ __launch_bounds__(256) void edge_msg_kernel(
    const int* __restrict__ esrc, const int* __restrict__ edst,
    const _Float16* __restrict__ hf16, const _Float16* __restrict__ eattr,
    const _Float16* __restrict__ we1f, const float* __restrict__ be1,
    const _Float16* __restrict__ we2f, const float* __restrict__ be2,
    float* __restrict__ agg) {
    __shared__ _Float16 w1s[36864];        // 9*8 frags * 512
    __shared__ _Float16 w2s[16384];        // 4*8 frags * 512
    __shared__ _Float16 m1s[8][16 * cH];   // per-wave stage-1 output

    // ---- async-stage all weight fragments into LDS ----
    for (int c = threadIdx.x; c < 6656; c += 256) {     // 16B chunks: 4608 + 2048
        if (c < 4608) async_copy_b128(w1s + c * 8, we1f + (size_t)c * 8);
        else          async_copy_b128(w2s + (c - 4608) * 8, we2f + (size_t)(c - 4608) * 8);
    }
    async_wait0();
    __syncthreads();

    const int wave = threadIdx.x >> 5;
    const int lane = threadIdx.x & 31;
    const int row  = lane & 15;
    const int hi   = lane >> 4;
    const int col0 = lane & 15;
    _Float16* mrow = m1s[wave];

    for (int t = blockIdx.x * 8 + wave; t < cE / 16; t += EDGE_GRID * 8) {
        const int e0 = t * 16;
        const int er = e0 + row;
        const int se = esrc[er], de = edst[er];
        const _Float16* ps = hf16 + (size_t)se * cH;
        const _Float16* pd = hf16 + (size_t)de * cH;
        const _Float16* pe = eattr + (size_t)er * 32;

        // stage 1: (16,288) x (288,128); K tiles: 4 h_src + 4 h_dst + 1 eattr
        v8f acc[8];
#pragma unroll
        for (int i = 0; i < 8; ++i) acc[i] = v8f_zero();

        for (int kt = 0; kt < 9; ++kt) {
            const _Float16* p = (kt < 4) ? (ps + kt * 32)
                              : (kt < 8) ? (pd + (kt - 4) * 32) : pe;
            v16h a = load_a_frag(p, hi);
            wmma_kstep(w1s + (size_t)kt * 8 * 512 + lane * 16, a, acc);
        }

        // bias + SiLU -> wave-private LDS tile (same-wave LDS ops are in-order)
#pragma unroll
        for (int nt = 0; nt < 8; ++nt) {
            float bb = be1[nt * 16 + col0];
#pragma unroll
            for (int j = 0; j < 8; ++j) {
                float x = silu_f(acc[nt][j] + bb);
                mrow[(j + hi * 8) * cH + nt * 16 + col0] = (_Float16)x;
            }
        }

        // stage 2: (16,128) x (128,128) from LDS
        v8f d2[8];
#pragma unroll
        for (int i = 0; i < 8; ++i) d2[i] = v8f_zero();

        for (int kt = 0; kt < 4; ++kt) {
            v16h a = load_a_frag(mrow + row * cH + kt * 32, hi);
            wmma_kstep(w2s + (size_t)kt * 8 * 512 + lane * 16, a, d2);
        }

        // bias + SiLU + scatter-add into agg[dst]
        int dsts[8];
#pragma unroll
        for (int j = 0; j < 8; ++j) dsts[j] = edst[e0 + j + hi * 8];
#pragma unroll
        for (int nt = 0; nt < 8; ++nt) {
            float bb = be2[nt * 16 + col0];
#pragma unroll
            for (int j = 0; j < 8; ++j) {
                float x = silu_f(d2[nt][j] + bb);
                atomicAdd(agg + (size_t)dsts[j] * cH + nt * 16 + col0, x);
            }
        }
    }
}

// ---------------- node update MLP ----------------
constexpr int NODE_GRID = 176;     // 176 blocks * 8 waves = 1408 waves; 2816 tiles = 2 per wave
__global__ __launch_bounds__(256) void node_mlp_kernel(
    const float* __restrict__ agg,
    const _Float16* __restrict__ wh1f, const float* __restrict__ bh1,
    const _Float16* __restrict__ wh2f, const float* __restrict__ bh2,
    float* __restrict__ h32, _Float16* __restrict__ hf16) {
    __shared__ _Float16 w1s[32768];        // 8*8 frags * 512
    __shared__ _Float16 w2s[16384];        // 4*8 frags * 512
    __shared__ _Float16 m1s[8][16 * cH];

    for (int c = threadIdx.x; c < 6144; c += 256) {     // 4096 + 2048 chunks
        if (c < 4096) async_copy_b128(w1s + c * 8, wh1f + (size_t)c * 8);
        else          async_copy_b128(w2s + (c - 4096) * 8, wh2f + (size_t)(c - 4096) * 8);
    }
    async_wait0();
    __syncthreads();

    const int wave = threadIdx.x >> 5;
    const int lane = threadIdx.x & 31;
    const int row  = lane & 15;
    const int hi   = lane >> 4;
    const int col0 = lane & 15;
    _Float16* mrow = m1s[wave];

    for (int t = blockIdx.x * 8 + wave; t < cA / 16; t += NODE_GRID * 8) {
        const int a0 = t * 16;
        const int ar = a0 + row;
        const _Float16* ph = hf16 + (size_t)ar * cH;
        const float*    pa = agg  + (size_t)ar * cH;

        // stage 1: (16,256) x (256,128); K tiles: 4 h + 4 agg(f32->f16)
        v8f acc[8];
#pragma unroll
        for (int i = 0; i < 8; ++i) acc[i] = v8f_zero();

        for (int kt = 0; kt < 8; ++kt) {
            v16h a;
            if (kt < 4) {
                a = load_a_frag(ph + kt * 32, hi);
            } else {
                const float* pq = pa + (kt - 4) * 32;
                float4 f0 = *(const float4*)(pq + hi * 8);
                float4 f1 = *(const float4*)(pq + hi * 8 + 4);
                float4 f2 = *(const float4*)(pq + 16 + hi * 8);
                float4 f3 = *(const float4*)(pq + 16 + hi * 8 + 4);
                a[0] = (_Float16)f0.x;  a[1] = (_Float16)f0.y;
                a[2] = (_Float16)f0.z;  a[3] = (_Float16)f0.w;
                a[4] = (_Float16)f1.x;  a[5] = (_Float16)f1.y;
                a[6] = (_Float16)f1.z;  a[7] = (_Float16)f1.w;
                a[8] = (_Float16)f2.x;  a[9] = (_Float16)f2.y;
                a[10] = (_Float16)f2.z; a[11] = (_Float16)f2.w;
                a[12] = (_Float16)f3.x; a[13] = (_Float16)f3.y;
                a[14] = (_Float16)f3.z; a[15] = (_Float16)f3.w;
            }
            wmma_kstep(w1s + (size_t)kt * 8 * 512 + lane * 16, a, acc);
        }

#pragma unroll
        for (int nt = 0; nt < 8; ++nt) {
            float bb = bh1[nt * 16 + col0];
#pragma unroll
            for (int j = 0; j < 8; ++j) {
                float x = silu_f(acc[nt][j] + bb);
                mrow[(j + hi * 8) * cH + nt * 16 + col0] = (_Float16)x;
            }
        }

        // stage 2: (16,128) x (128,128), bias (no SiLU), residual add
        v8f d2[8];
#pragma unroll
        for (int i = 0; i < 8; ++i) d2[i] = v8f_zero();

        for (int kt = 0; kt < 4; ++kt) {
            v16h a = load_a_frag(mrow + row * cH + kt * 32, hi);
            wmma_kstep(w2s + (size_t)kt * 8 * 512 + lane * 16, a, d2);
        }

#pragma unroll
        for (int nt = 0; nt < 8; ++nt) {
            float bb = bh2[nt * 16 + col0];
#pragma unroll
            for (int j = 0; j < 8; ++j) {
                size_t idx = (size_t)(a0 + j + hi * 8) * cH + nt * 16 + col0;
                float nh = h32[idx] + d2[nt][j] + bb;
                h32[idx] = nh;
                hf16[idx] = (_Float16)nh;
            }
        }
    }
}

// ---------------- (A,128) x (128,128) + bias projection (k / v) ----------------
__global__ __launch_bounds__(256) void lin128_kernel(
    const _Float16* __restrict__ hf16, const _Float16* __restrict__ wf,
    const float* __restrict__ bias, float* __restrict__ out) {
    __shared__ _Float16 ws_[16384];        // 4*8 frags * 512

    for (int c = threadIdx.x; c < 2048; c += 256)
        async_copy_b128(ws_ + c * 8, wf + (size_t)c * 8);
    async_wait0();
    __syncthreads();

    const int wave = threadIdx.x >> 5;
    const int lane = threadIdx.x & 31;
    const int row  = lane & 15;
    const int hi   = lane >> 4;
    const int col0 = lane & 15;

    for (int t = blockIdx.x * 8 + wave; t < cA / 16; t += NODE_GRID * 8) {
        const int a0 = t * 16;
        const _Float16* ph = hf16 + (size_t)(a0 + row) * cH;

        v8f acc[8];
#pragma unroll
        for (int i = 0; i < 8; ++i) acc[i] = v8f_zero();

        for (int kt = 0; kt < 4; ++kt) {
            v16h a = load_a_frag(ph + kt * 32, hi);
            wmma_kstep(ws_ + (size_t)kt * 8 * 512 + lane * 16, a, acc);
        }
#pragma unroll
        for (int nt = 0; nt < 8; ++nt) {
            float bb = bias[nt * 16 + col0];
#pragma unroll
            for (int j = 0; j < 8; ++j) {
                out[(size_t)(a0 + j + hi * 8) * cH + nt * 16 + col0] = acc[nt][j] + bb;
            }
        }
    }
}

// ---------------- attention pooling pieces ----------------
__global__ void q_kernel(const int* __restrict__ rt, const float* __restrict__ remb,
                         const float* __restrict__ Wq, const float* __restrict__ bq,
                         float* __restrict__ q32) {
    int idx = blockIdx.x * blockDim.x + threadIdx.x;   // N*H exact
    int n = idx >> 7, c = idx & 127;
    const float* e = remb + (size_t)rt[n] * cH;
    float s = bq[c];
    for (int k = 0; k < cH; ++k) s += e[k] * Wq[(size_t)k * cH + c];
    q32[idx] = s;
}

__global__ void scores_kernel(const float* __restrict__ q32, const float* __restrict__ k32,
                              const int* __restrict__ ri,
                              float* __restrict__ scores, float* __restrict__ smax) {
    int a = blockIdx.x * blockDim.x + threadIdx.x;     // A exact
    int r = ri[a];
    const float* qp = q32 + (size_t)r * cH;
    const float* kp = k32 + (size_t)a * cH;
    float s = 0.f;
#pragma unroll 8
    for (int c = 0; c < cH; ++c) s += qp[c] * kp[c];
    s *= 0.08838834764831845f;                          // H^-0.5
    scores[a] = s;
    atomicMaxF(smax + r, s);
}

__global__ void expsum_kernel(const int* __restrict__ ri, const float* __restrict__ smax,
                              float* __restrict__ scores, float* __restrict__ denom) {
    int a = blockIdx.x * blockDim.x + threadIdx.x;
    int r = ri[a];
    float ex = __expf(scores[a] - smax[r]);
    scores[a] = ex;
    atomicAdd(denom + r, ex);
}

__global__ void pooled_kernel(const int* __restrict__ ri, const float* __restrict__ ex,
                              const float* __restrict__ denom, const float* __restrict__ v32,
                              float* __restrict__ pooled) {
    int idx = blockIdx.x * blockDim.x + threadIdx.x;   // A*H exact
    int a = idx >> 7, c = idx & 127;
    int r = ri[a];
    float w = ex[a] / denom[r];
    atomicAdd(pooled + (size_t)r * cH + c, w * v32[idx]);
}

__global__ void out_kernel(const float* __restrict__ pooled,
                           const float* __restrict__ Wmu, const float* __restrict__ bmu,
                           const float* __restrict__ Wlv, const float* __restrict__ blv,
                           float* __restrict__ out) {
    int idx = blockIdx.x * blockDim.x + threadIdx.x;   // N*LAT exact
    int n = idx >> 5, j = idx & 31;
    const float* pp = pooled + (size_t)n * cH;
    float m = bmu[j], lv = blv[j];
    for (int c = 0; c < cH; ++c) {
        m  += pp[c] * Wmu[(size_t)c * cLAT + j];
        lv += pp[c] * Wlv[(size_t)c * cLAT + j];
    }
    lv = fminf(fmaxf(lv, -10.f), 2.f);
    out[idx] = m;
    out[(size_t)cN * cLAT + idx] = lv;
}

// ---------------- host driver ----------------
extern "C" void kernel_launch(void* const* d_in, const int* in_sizes, int n_in,
                              void* d_out, int out_size, void* d_ws, size_t ws_size,
                              hipStream_t stream) {
    const float* atom_coords     = (const float*)d_in[0];
    const int*   atom_types      = (const int*)d_in[1];
    const int*   residue_indices = (const int*)d_in[2];
    const int*   residue_types   = (const int*)d_in[3];
    const int*   edge_src        = (const int*)d_in[5];
    const int*   edge_dst        = (const int*)d_in[6];
    const float* atom_embed      = (const float*)d_in[7];
    const float* residue_embed   = (const float*)d_in[8];
    const float* We1 = (const float*)d_in[9];
    const float* be1 = (const float*)d_in[10];
    const float* We2 = (const float*)d_in[11];
    const float* be2 = (const float*)d_in[12];
    const float* Wh1 = (const float*)d_in[13];
    const float* bh1 = (const float*)d_in[14];
    const float* Wh2 = (const float*)d_in[15];
    const float* bh2 = (const float*)d_in[16];
    const float* Wq  = (const float*)d_in[17];
    const float* bq  = (const float*)d_in[18];
    const float* Wk  = (const float*)d_in[19];
    const float* bk  = (const float*)d_in[20];
    const float* Wv  = (const float*)d_in[21];
    const float* bv  = (const float*)d_in[22];
    const float* Wmu = (const float*)d_in[23];
    const float* bmu = (const float*)d_in[24];
    const float* Wlv = (const float*)d_in[25];
    const float* blv = (const float*)d_in[26];

    char* ws = (char*)d_ws;
    float*     h32    = (float*)(ws + OFF_H32);
    _Float16*  hf16   = (_Float16*)(ws + OFF_HF16);
    float*     agg    = (float*)(ws + OFF_AGG);
    _Float16*  eattr  = (_Float16*)(ws + OFF_EATTR);
    float*     q32    = (float*)(ws + OFF_Q32);
    float*     scores = (float*)(ws + OFF_SCORES);
    float*     smax   = (float*)(ws + OFF_SMAX);
    float*     denom  = (float*)(ws + OFF_DENOM);
    float*     pooled = (float*)(ws + OFF_POOLED);
    _Float16*  wfrag  = (_Float16*)(ws + OFF_WFRAG);
    float*     k32    = agg;                      // reuse agg after last layer
    float*     v32    = (float*)(ws + OFF_EATTR); // reuse eattr after last layer
    float*     out    = (float*)d_out;

    // weight repack into WMMA B-fragments (f16)
    for (int l = 0; l < 3; ++l) {
        repack_kernel<<<9 * 16, 256, 0, stream>>>(We1 + (size_t)l * 272 * cH, wfrag + WF_WE1 + (size_t)l * 36864, 272, 9);
        repack_kernel<<<4 * 16, 256, 0, stream>>>(We2 + (size_t)l * cH * cH,  wfrag + WF_WE2 + (size_t)l * 16384, 128, 4);
        repack_kernel<<<8 * 16, 256, 0, stream>>>(Wh1 + (size_t)l * 256 * cH, wfrag + WF_WH1 + (size_t)l * 32768, 256, 8);
        repack_kernel<<<4 * 16, 256, 0, stream>>>(Wh2 + (size_t)l * cH * cH,  wfrag + WF_WH2 + (size_t)l * 16384, 128, 4);
    }
    repack_kernel<<<64, 256, 0, stream>>>(Wk, wfrag + WF_WK, 128, 4);
    repack_kernel<<<64, 256, 0, stream>>>(Wv, wfrag + WF_WV, 128, 4);

    // edge RBF features (centering cancels in pairwise distances) + node init
    rbf_kernel<<<cE / 256, 256, 0, stream>>>(atom_coords, edge_src, edge_dst, eattr);
    hinit_kernel<<<cA * cH / 256, 256, 0, stream>>>(atom_types, residue_indices, residue_types,
                                                    atom_embed, residue_embed, h32, hf16);

    // 3 EGNN layers
    for (int l = 0; l < 3; ++l) {
        fill_kernel<<<cA * cH / 256, 256, 0, stream>>>(agg, 0.f, cA * cH);
        edge_msg_kernel<<<EDGE_GRID, 256, 0, stream>>>(edge_src, edge_dst, hf16, eattr,
                                                       wfrag + WF_WE1 + (size_t)l * 36864, be1 + l * cH,
                                                       wfrag + WF_WE2 + (size_t)l * 16384, be2 + l * cH, agg);
        node_mlp_kernel<<<NODE_GRID, 256, 0, stream>>>(agg,
                                                       wfrag + WF_WH1 + (size_t)l * 32768, bh1 + l * cH,
                                                       wfrag + WF_WH2 + (size_t)l * 16384, bh2 + l * cH,
                                                       h32, hf16);
    }

    // attention pooling
    q_kernel<<<cN * cH / 256, 256, 0, stream>>>(residue_types, residue_embed, Wq, bq, q32);
    lin128_kernel<<<NODE_GRID, 256, 0, stream>>>(hf16, wfrag + WF_WK, bk, k32);
    lin128_kernel<<<NODE_GRID, 256, 0, stream>>>(hf16, wfrag + WF_WV, bv, v32);

    fill_kernel<<<(cN + 255) / 256, 256, 0, stream>>>(smax, -1e30f, cN);
    fill_kernel<<<(cN + 255) / 256, 256, 0, stream>>>(denom, 0.f, cN);
    fill_kernel<<<cN * cH / 256, 256, 0, stream>>>(pooled, 0.f, cN * cH);

    scores_kernel<<<cA / 256, 256, 0, stream>>>(q32, k32, residue_indices, scores, smax);
    expsum_kernel<<<cA / 256, 256, 0, stream>>>(residue_indices, smax, scores, denom);
    pooled_kernel<<<cA * cH / 256, 256, 0, stream>>>(residue_indices, scores, denom, v32, pooled);

    // output heads
    out_kernel<<<cN * cLAT / 256, 256, 0, stream>>>(pooled, Wmu, bmu, Wlv, blv, out);
}
